// Pyramid_3770981286106
// MI455X (gfx1250) — matrix operand order repaired and checked by reading, not compiled, not run on hardware
//
#include <hip/hip_runtime.h>
#include <math.h>

typedef _Float16 h16;
typedef __attribute__((ext_vector_type(8)))  _Float16 v8h;
typedef __attribute__((ext_vector_type(16))) _Float16 v16h;
typedef __attribute__((ext_vector_type(8)))  float    v8f;

#define B_SZ   8
#define SEQL   1000
#define C_IN   32
#define DIM    512
#define NHEAD  8
#define DKH    64
#define DFF    2048
#define NLAYER 4
#define LEAF   1024
#define NTOT   1365
#define NROWS  (B_SZ * NTOT)   /* 10920 */
#define NEFF   1334

__constant__ int c_offs[7] = {0, 1024, 1280, 1344, 1360, 1364, 1365};
__constant__ int c_size[6] = {1024, 256, 64, 16, 4, 1};
__constant__ int c_eoff[7] = {0, 1000, 1250, 1313, 1329, 1333, 1334};

// ---------------- embedding: circular conv1d(k=3) + sinusoid PE, pad to 1024 ----
__global__ void conv_pe_kernel(const float* __restrict__ x,
                               const float* __restrict__ w,
                               float* __restrict__ emb) {
  int idx = blockIdx.x * blockDim.x + threadIdx.x;
  if (idx >= B_SZ * LEAF * DIM) return;
  int d = idx % DIM;
  int s = (idx / DIM) % LEAF;
  int b = idx / (DIM * LEAF);
  float acc = 0.0f;
  if (s < SEQL) {
    #pragma unroll
    for (int t = 0; t < 3; ++t) {
      int sr = (s - 1 + t + SEQL) % SEQL;
      const float* xp = x + (size_t)(b * SEQL + sr) * C_IN;
      const float* wp = w + (size_t)d * C_IN * 3 + t;
      #pragma unroll 8
      for (int c = 0; c < C_IN; ++c) acc += xp[c] * wp[c * 3];
    }
    float e   = (float)(2 * (d >> 1)) * (1.0f / (float)DIM);
    float inv = __expf(-9.210340371976184f * e); // 10000^-e
    float ang = (float)s * inv;
    acc += (d & 1) ? cosf(ang) : sinf(ang);
  }
  emb[idx] = acc;
}

// ---------------- hierarchical max-pool: node (l,i) = max over its 4^l leaves ---
__global__ void pool_kernel(const float* __restrict__ emb, float* __restrict__ seq) {
  int idx = blockIdx.x * blockDim.x + threadIdx.x;
  if (idx >= NROWS * DIM) return;
  int d = idx % DIM;
  int n = (idx / DIM) % NTOT;
  int b = idx / (DIM * NTOT);
  int l = 0;
  while (n >= c_offs[l + 1]) ++l;
  int i = n - c_offs[l];
  int cnt = 1 << (2 * l);
  const float* ep = emb + ((size_t)b * LEAF + (size_t)i * cnt) * DIM + d;
  float m = ep[0];
  for (int j = 1; j < cnt; ++j) m = fmaxf(m, ep[(size_t)j * DIM]);
  seq[idx] = m;
}

// ---------------- LayerNorm over D=512, fused fp32 + fp16 outputs --------------
__global__ void ln_kernel(const float* __restrict__ x, const float* __restrict__ g,
                          const float* __restrict__ bt, float* __restrict__ y,
                          h16* __restrict__ yh) {
  int row = blockIdx.x;
  int t = threadIdx.x; // 128 threads
  const float* xr = x + (size_t)row * DIM;
  float v[4];
  float s = 0.0f;
  #pragma unroll
  for (int i = 0; i < 4; ++i) { v[i] = xr[t + i * 128]; s += v[i]; }
  __shared__ float red[128];
  red[t] = s;
  __syncthreads();
  for (int st = 64; st > 0; st >>= 1) { if (t < st) red[t] += red[t + st]; __syncthreads(); }
  float mean = red[0] * (1.0f / DIM);
  __syncthreads();
  float q = 0.0f;
  #pragma unroll
  for (int i = 0; i < 4; ++i) { float dd = v[i] - mean; q += dd * dd; }
  red[t] = q;
  __syncthreads();
  for (int st = 64; st > 0; st >>= 1) { if (t < st) red[t] += red[t + st]; __syncthreads(); }
  float rstd = rsqrtf(red[0] * (1.0f / DIM) + 1e-5f);
  float* yr = y + (size_t)row * DIM;
  h16* yhr = yh + (size_t)row * DIM;
  #pragma unroll
  for (int i = 0; i < 4; ++i) {
    int d = t + i * 128;
    float o = (v[i] - mean) * rstd * g[d] + bt[d];
    yr[d] = o;
    yhr[d] = (h16)o;
  }
}

// ---------------- f32 -> f16 with column placement (for packed QKV weights) ----
__global__ void f2h_cols(const float* __restrict__ src, h16* __restrict__ dst,
                         int rows, int cols, int ldd, int col0) {
  int idx = blockIdx.x * blockDim.x + threadIdx.x;
  if (idx >= rows * cols) return;
  int r = idx / cols, c = idx % cols;
  dst[(size_t)r * ldd + col0 + c] = (h16)src[idx];
}

// ---------------- WMMA fp16->fp32 GEMM: C = A[M,K] * B[K,N] (+bias,+res,gelu) --
// 128 threads = 4 waves; block tile 64(M) x 128(N); wave tile 32x64 = 2x4 wmma.
// B tile is staged in LDS pre-swizzled into the wave32 B-fragment layout so each
// lane's v16h fragment is one contiguous 32-byte LDS read (2x ds_load_b128).
__global__ __launch_bounds__(128)
void gemm_kernel(const h16* __restrict__ A, const h16* __restrict__ Bw,
                 const float* __restrict__ bias, const float* __restrict__ resid,
                 float* __restrict__ Cf, h16* __restrict__ Ch,
                 int M, int N, int K, int act) {
  __shared__ h16 As[64][40];         // 64 rows x 32 K (pad to 40)
  __shared__ h16 Bsw[8][32][16];     // [colgrp][lane][kidx] fragment-order B tile
  int tid  = threadIdx.x;
  int lane = tid & 31, wave = tid >> 5;
  int wm = (wave >> 1) * 32;         // wave row base   (0 / 32)
  int wn = (wave & 1) * 64;          // wave col base   (0 / 64)
  int hh = lane >> 4, lm = lane & 15;
  int by = blockIdx.y, bx = blockIdx.x;

  v8f acc[2][4] = {};

  for (int k0 = 0; k0 < K; k0 += 32) {
    // --- A tile: 64x32 halves as 16-byte chunks, zero-fill rows >= M ---
    #pragma unroll
    for (int e = tid; e < 256; e += 128) {
      int r = e >> 2, c8 = (e & 3) * 8;
      int gm = by * 64 + r;
      int4 val = {0, 0, 0, 0};
      if (gm < M) val = *(const int4*)(A + (size_t)gm * K + k0 + c8);
      *(int4*)&As[r][c8] = val;
    }
    // --- B tile: 32(K) x 128(N), vector-load rows, scatter into swizzle ---
    #pragma unroll
    for (int e = tid; e < 512; e += 128) {
      int k = e >> 4, c8 = (e & 15) * 8;
      v8h vb = *(const v8h*)(Bw + (size_t)(k0 + k) * N + bx * 128 + c8);
      int kh = (k >> 3) & 1, kidx = (k & 7) | ((k >> 4) << 3);
      #pragma unroll
      for (int j = 0; j < 8; ++j) {
        int c = c8 + j;
        Bsw[c >> 4][kh * 16 + (c & 15)][kidx] = vb[j];
      }
    }
    __syncthreads();

    // --- fragments ---
    v16h af[2], bf[4];
    #pragma unroll
    for (int ti = 0; ti < 2; ++ti) {
      int r = wm + ti * 16 + lm;
      v8h lo = *(const v8h*)&As[r][hh * 8];        // K = hh*8 .. +7
      v8h hi = *(const v8h*)&As[r][16 + hh * 8];   // K = 16+hh*8 .. +7
      af[ti] = __builtin_shufflevector(lo, hi, 0, 1, 2, 3, 4, 5, 6, 7,
                                       8, 9, 10, 11, 12, 13, 14, 15);
    }
    #pragma unroll
    for (int tj = 0; tj < 4; ++tj) {
      int cg = (wn >> 4) + tj;
      bf[tj] = *(const v16h*)&Bsw[cg][lane][0];
    }
    #pragma unroll
    for (int ti = 0; ti < 2; ++ti)
      #pragma unroll
      for (int tj = 0; tj < 4; ++tj)
        acc[ti][tj] = __builtin_amdgcn_wmma_f32_16x16x32_f16(
            false, af[ti], false, bf[tj], (short)0, acc[ti][tj], false, false);
    __syncthreads();
  }

  // --- epilogue: C/D layout — lane col = lm, VGPR r is row r + 8*hh ---
  #pragma unroll
  for (int ti = 0; ti < 2; ++ti)
    #pragma unroll
    for (int tj = 0; tj < 4; ++tj) {
      int ncol = bx * 128 + wn + tj * 16 + lm;
      #pragma unroll
      for (int r = 0; r < 8; ++r) {
        int mrow = by * 64 + wm + ti * 16 + hh * 8 + r;
        if (mrow < M) {
          float vv = acc[ti][tj][r];
          if (bias)  vv += bias[ncol];
          if (resid) vv += resid[(size_t)mrow * N + ncol];
          if (act == 1) vv = 0.5f * vv * (1.0f + erff(vv * 0.70710678118f));
          size_t o = (size_t)mrow * N + ncol;
          if (Cf) Cf[o] = vv;
          if (Ch) Ch[o] = (h16)vv;
        }
      }
    }
}

// ---------------- sparse pyramid attention: <=10 keys per query -----------------
__global__ void attn_kernel(const float* __restrict__ qkv, h16* __restrict__ ctx) {
  int idx = blockIdx.x * blockDim.x + threadIdx.x;
  if (idx >= NROWS * NHEAD) return;
  int h = idx % NHEAD;
  int n = (idx / NHEAD) % NTOT;
  int b = idx / (NHEAD * NTOT);
  int l = 0;
  while (n >= c_offs[l + 1]) ++l;
  int i = n - c_offs[l];
  int sz = c_size[l];
  int nb[10];
  int cnt = 0;
  int lo = (i - 2 > 0) ? i - 2 : 0;
  int hi = (i + 2 < sz - 1) ? i + 2 : sz - 1;
  for (int j = lo; j <= hi; ++j) nb[cnt++] = c_offs[l] + j;
  if (l >= 1) {
    int base = c_offs[l - 1] + i * 4;
    for (int c = 0; c < 4; ++c) nb[cnt++] = base + c;
  }
  if (l <= 4) nb[cnt++] = c_offs[l + 1] + (i >> 2);

  const float* qp = qkv + (size_t)(b * NTOT + n) * 1536 + h * DKH;
  float q[DKH];
  #pragma unroll
  for (int d = 0; d < DKH; ++d) q[d] = qp[d];

  float sc[10];
  float mx = -1e30f;
  for (int m = 0; m < cnt; ++m) {
    const float* kp = qkv + (size_t)(b * NTOT + nb[m]) * 1536 + 512 + h * DKH;
    float dot = 0.0f;
    #pragma unroll 16
    for (int d = 0; d < DKH; ++d) dot += q[d] * kp[d];
    sc[m] = dot * 0.125f; // 1/sqrt(64)
    mx = fmaxf(mx, sc[m]);
  }
  float sum = 0.0f;
  for (int m = 0; m < cnt; ++m) { sc[m] = __expf(sc[m] - mx); sum += sc[m]; }
  float inv = 1.0f / sum;

  float acc[DKH];
  #pragma unroll
  for (int d = 0; d < DKH; ++d) acc[d] = 0.0f;
  for (int m = 0; m < cnt; ++m) {
    const float* vp = qkv + (size_t)(b * NTOT + nb[m]) * 1536 + 1024 + h * DKH;
    float p = sc[m] * inv;
    #pragma unroll 16
    for (int d = 0; d < DKH; ++d) acc[d] += p * vp[d];
  }
  h16* cp = ctx + (size_t)(b * NTOT + n) * DIM + h * DKH;
  #pragma unroll
  for (int d = 0; d < DKH; ++d) cp[d] = (h16)acc[d];
}

// ---------------- final gather of effective prefixes ----------------------------
__global__ void gather_kernel(const float* __restrict__ seq, float* __restrict__ out) {
  int idx = blockIdx.x * blockDim.x + threadIdx.x;
  if (idx >= B_SZ * NEFF * DIM) return;
  int d = idx % DIM;
  int t = (idx / DIM) % NEFF;
  int b = idx / (DIM * NEFF);
  int s = 0;
  while (t >= c_eoff[s + 1]) ++s;
  int node = c_offs[s] + (t - c_eoff[s]);
  out[idx] = seq[(size_t)(b * NTOT + node) * DIM + d];
}

// ================================================================================
extern "C" void kernel_launch(void* const* d_in, const int* in_sizes, int n_in,
                              void* d_out, int out_size, void* d_ws, size_t ws_size,
                              hipStream_t stream) {
  (void)in_sizes; (void)n_in; (void)out_size; (void)ws_size;
  const float* x      = (const float*)d_in[0];
  const float* conv_w = (const float*)d_in[1];
  const float* norm_s = (const float*)d_in[2];
  const float* norm_b = (const float*)d_in[3];
  const float* wq     = (const float*)d_in[4];
  const float* wk     = (const float*)d_in[5];
  const float* wv     = (const float*)d_in[6];
  const float* wo     = (const float*)d_in[7];
  const float* ln1_s  = (const float*)d_in[8];
  const float* ln1_b  = (const float*)d_in[9];
  const float* w1     = (const float*)d_in[10];
  const float* b1     = (const float*)d_in[11];
  const float* w2     = (const float*)d_in[12];
  const float* b2     = (const float*)d_in[13];
  const float* ln2_s  = (const float*)d_in[14];
  const float* ln2_b  = (const float*)d_in[15];
  float* out = (float*)d_out;

  char* base = (char*)d_ws;
  size_t off = 0;
  auto carve = [&](size_t bytes) -> void* {
    void* p = base + off;
    off += (bytes + 255) & ~(size_t)255;
    return p;
  };
  float* emb    = (float*)carve((size_t)B_SZ * LEAF * DIM * 4);
  float* seq    = (float*)carve((size_t)NROWS * DIM * 4);
  h16*   seq_h  = (h16*)  carve((size_t)NROWS * DIM * 2);
  float* big    = (float*)carve((size_t)NROWS * 1536 * 4); // qkv, later residual-sum tmp
  h16*   ctx_h  = (h16*)  carve((size_t)NROWS * DIM * 2);
  h16*   ff_h   = (h16*)  carve((size_t)NROWS * DFF * 2);
  h16*   wqkv_h = (h16*)  carve((size_t)DIM * 1536 * 2);
  h16*   wo_h   = (h16*)  carve((size_t)DIM * DIM * 2);
  h16*   w1_h   = (h16*)  carve((size_t)DIM * DFF * 2);
  h16*   w2_h   = (h16*)  carve((size_t)DFF * DIM * 2);

  {
    int n = B_SZ * LEAF * DIM;
    conv_pe_kernel<<<(n + 255) / 256, 256, 0, stream>>>(x, conv_w, emb);
  }
  {
    int n = NROWS * DIM;
    pool_kernel<<<(n + 255) / 256, 256, 0, stream>>>(emb, seq);
  }
  ln_kernel<<<NROWS, 128, 0, stream>>>(seq, norm_s, norm_b, seq, seq_h);

  dim3 blk(128);
  int gy = (NROWS + 63) / 64;
  for (int l = 0; l < NLAYER; ++l) {
    {
      int n = DIM * DIM;
      int g = (n + 255) / 256;
      f2h_cols<<<g, 256, 0, stream>>>(wq + (size_t)l * DIM * DIM, wqkv_h, DIM, DIM, 1536, 0);
      f2h_cols<<<g, 256, 0, stream>>>(wk + (size_t)l * DIM * DIM, wqkv_h, DIM, DIM, 1536, 512);
      f2h_cols<<<g, 256, 0, stream>>>(wv + (size_t)l * DIM * DIM, wqkv_h, DIM, DIM, 1536, 1024);
      f2h_cols<<<g, 256, 0, stream>>>(wo + (size_t)l * DIM * DIM, wo_h, DIM, DIM, DIM, 0);
    }
    {
      int n = DIM * DFF;
      int g = (n + 255) / 256;
      f2h_cols<<<g, 256, 0, stream>>>(w1 + (size_t)l * DIM * DFF, w1_h, DIM, DFF, DFF, 0);
      f2h_cols<<<g, 256, 0, stream>>>(w2 + (size_t)l * DFF * DIM, w2_h, DFF, DIM, DIM, 0);
    }
    // QKV: [10920,512] x [512,1536]
    gemm_kernel<<<dim3(1536 / 128, gy), blk, 0, stream>>>(
        seq_h, wqkv_h, nullptr, nullptr, big, nullptr, NROWS, 1536, DIM, 0);
    {
      int n = NROWS * NHEAD;
      attn_kernel<<<(n + 255) / 256, 256, 0, stream>>>(big, ctx_h);
    }
    // O-proj + residual -> big (fp32), then LN1 -> seq/seq_h
    gemm_kernel<<<dim3(DIM / 128, gy), blk, 0, stream>>>(
        ctx_h, wo_h, nullptr, seq, big, nullptr, NROWS, DIM, DIM, 0);
    ln_kernel<<<NROWS, 128, 0, stream>>>(big, ln1_s + l * DIM, ln1_b + l * DIM, seq, seq_h);
    // FFN1 + bias + exact GELU -> fp16 directly
    gemm_kernel<<<dim3(DFF / 128, gy), blk, 0, stream>>>(
        seq_h, w1_h, b1 + (size_t)l * DFF, nullptr, nullptr, ff_h, NROWS, DFF, DIM, 1);
    // FFN2 + bias + residual -> big (fp32), then LN2 -> seq/seq_h
    gemm_kernel<<<dim3(DIM / 128, gy), blk, 0, stream>>>(
        ff_h, w2_h, b2 + (size_t)l * DIM, seq, big, nullptr, NROWS, DIM, DFF, 0);
    ln_kernel<<<NROWS, 128, 0, stream>>>(big, ln2_s + l * DIM, ln2_b + l * DIM, seq, seq_h);
  }
  {
    int n = B_SZ * NEFF * DIM;
    gather_kernel<<<(n + 255) / 256, 256, 0, stream>>>(seq, out);
  }
}
